// GraphTopologicalLoss_81003083202863
// MI455X (gfx1250) — compile-verified, matrix operand order checked by reading, not run on hardware
//
#include <hip/hip_runtime.h>
#include <hip/hip_bf16.h>

typedef __attribute__((ext_vector_type(8))) int v8i;

#define N_NODES   4096
#define MAT_BYTES ((size_t)N_NODES * N_NODES)   // 16 MiB byte adjacency
#define TILE_B    1024                          // bytes per IU8 WMMA (16x64 u8 A-tile)
#define TILES_PER_WAVE 16                       // accumulate 16 tiles into one C

// ---------------------------------------------------------------------------
// Phase 1: zero the byte adjacency matrix (uint4 stores) + zero the counters.
// ---------------------------------------------------------------------------
__global__ void gtl_zero_kernel(uint4* __restrict__ mat4,
                                unsigned* __restrict__ counters, int n4) {
    int i = blockIdx.x * blockDim.x + threadIdx.x;
    if (i < n4) mat4[i] = make_uint4(0u, 0u, 0u, 0u);
    if (i == 0) { counters[0] = 0u; counters[1] = 0u; }
}

// ---------------------------------------------------------------------------
// Phase 2: scatter edges. Stored value is always 1 -> idempotent plain byte
// stores; duplicates and the symmetric pair need no atomics. Random access is
// absorbed by the 192 MB L2 (whole matrix resident).
// ---------------------------------------------------------------------------
__global__ void gtl_scatter_kernel(const int* __restrict__ ei,
                                   unsigned char* __restrict__ adj, int E) {
    int e = blockIdx.x * blockDim.x + threadIdx.x;
    if (e >= E) return;
    int a = ei[e]     & (N_NODES - 1);
    int b = ei[E + e] & (N_NODES - 1);
    adj[(size_t)a * N_NODES + b] = (unsigned char)1;
    adj[(size_t)b * N_NODES + a] = (unsigned char)1;
}

// ---------------------------------------------------------------------------
// Phase 3: trace = count of diagonal ones (expected ~E/N nonzero, cheap).
// ---------------------------------------------------------------------------
__global__ void gtl_trace_kernel(const unsigned char* __restrict__ adj,
                                 unsigned* __restrict__ counters) {
    int i = blockIdx.x * blockDim.x + threadIdx.x;
    if (i < N_NODES) {
        unsigned v = adj[(size_t)i * N_NODES + i];
        if (v) atomicAdd(&counters[1], v);
    }
}

// ---------------------------------------------------------------------------
// Phase 4: S = total nonzero count via chained V_WMMA_I32_16X16X64_IU8.
// Each wave streams 16 KiB (16 tiles of 1 KiB); each lane loads 32 contiguous
// bytes per tile as the A fragment. Byte->fragment permutation is irrelevant
// because B is all-ones: the sum of all D entries is always 16 * sum(A-tile).
// C accumulates across the 16 tiles (D->C chaining), then one ds_swizzle
// butterfly + one atomic per wave.
// ---------------------------------------------------------------------------
__global__ void gtl_wmma_reduce_kernel(const unsigned char* __restrict__ adj,
                                       unsigned* __restrict__ counters) {
    int flat = blockIdx.x * blockDim.x + threadIdx.x;
    int wave = flat >> 5;
    int lane = threadIdx.x & 31;

    const unsigned char* base =
        adj + (size_t)wave * (TILE_B * TILES_PER_WAVE) + (size_t)lane * 32;

    v8i b;                            // B = all-ones u8 matrix (64x16)
#pragma unroll
    for (int k = 0; k < 8; ++k) b[k] = 0x01010101;

    v8i c = {};                       // i32 accumulator, chained across tiles
#pragma unroll
    for (int t = 0; t < TILES_PER_WAVE; ++t) {
        v8i a = *(const v8i*)(base + (size_t)t * TILE_B);
        // 7 args: (sgn_a, A, sgn_b, B, C, reuse_a, reuse_b) — unsigned x unsigned
        c = __builtin_amdgcn_wmma_i32_16x16x64_iu8(false, a, false, b, c,
                                                   false, false);
    }

    int s = 0;
#pragma unroll
    for (int k = 0; k < 8; ++k) s += c[k];

    // wave32 XOR butterfly via ds_swizzle group-of-32 patterns:
    // offset = (xor_mask << 10) | 0x1F  (and_mask=0x1F, or_mask=0)
    s += __builtin_amdgcn_ds_swizzle(s, 0x401F);  // SWAPX16
    s += __builtin_amdgcn_ds_swizzle(s, 0x201F);  // SWAPX8
    s += __builtin_amdgcn_ds_swizzle(s, 0x101F);  // SWAPX4
    s += __builtin_amdgcn_ds_swizzle(s, 0x081F);  // SWAPX2
    s += __builtin_amdgcn_ds_swizzle(s, 0x041F);  // SWAPX1

    // every lane now holds sum over all 16 columns = 16 * segment sum
    if (lane == 0) atomicAdd(&counters[0], (unsigned)(s >> 4));
}

// ---------------------------------------------------------------------------
// Phase 5: var(eig) = (S - T^2/n) / (n-1); double for the 2^24-range sum.
// ---------------------------------------------------------------------------
__global__ void gtl_finalize_kernel(const unsigned* __restrict__ counters,
                                    float* __restrict__ out) {
    double S = (double)counters[0];
    double T = (double)counters[1];
    double n = (double)N_NODES;
    out[0] = (float)((S - T * T / n) / (n - 1.0));
}

extern "C" void kernel_launch(void* const* d_in, const int* in_sizes, int n_in,
                              void* d_out, int out_size, void* d_ws, size_t ws_size,
                              hipStream_t stream) {
    (void)n_in; (void)out_size; (void)ws_size;
    const int* ei = (const int*)d_in[1];          // edge_index, flat (2, E) row-major
    int E = in_sizes[1] / 2;

    unsigned*      counters = (unsigned*)d_ws;               // [0]=S, [1]=T
    unsigned char* adj      = (unsigned char*)d_ws + 64;     // 16 MiB, 32B-aligned
    float*         out      = (float*)d_out;

    int n4 = (int)(MAT_BYTES / 16);
    gtl_zero_kernel<<<(n4 + 255) / 256, 256, 0, stream>>>((uint4*)adj, counters, n4);

    gtl_scatter_kernel<<<(E + 255) / 256, 256, 0, stream>>>(ei, adj, E);

    gtl_trace_kernel<<<(N_NODES + 255) / 256, 256, 0, stream>>>(adj, counters);

    int waves  = (int)(MAT_BYTES / (TILE_B * TILES_PER_WAVE));  // 1024 waves
    int blocks = waves * 32 / 256;                              // 128 blocks
    gtl_wmma_reduce_kernel<<<blocks, 256, 0, stream>>>(adj, counters);

    gtl_finalize_kernel<<<1, 1, 0, stream>>>(counters, out);
}
